// CustomGIN_34041910788208
// MI455X (gfx1250) — compile-verified, matrix operand order checked by reading, not run on hardware
//
#include <hip/hip_runtime.h>

#define NN   200000
#define EE   600000
#define GG   256
#define HH   128
#define OUTD 7
#define LL   12
#define WSTR 136   // LDS row stride in halfwords: 272B = 17*16B -> b128-aligned, conflict-free

typedef __bf16        v16bf __attribute__((ext_vector_type(16)));
typedef float         v8f   __attribute__((ext_vector_type(8)));
typedef unsigned int  v8u   __attribute__((ext_vector_type(8)));

// ---- fp32 -> bf16 (round-to-nearest-even) ----
__device__ __forceinline__ unsigned short f2bf(float f) {
  unsigned int u = __builtin_bit_cast(unsigned int, f);
  unsigned int r = u + 0x7fffu + ((u >> 16) & 1u);
  return (unsigned short)(r >> 16);
}
__device__ __forceinline__ unsigned int pack2bf(float a, float b) {
  return (unsigned int)f2bf(a) | ((unsigned int)f2bf(b) << 16);
}

// ---------------- small utility kernels ----------------
__global__ void k_fill_zero(float* __restrict__ p, int n) {
  int i = blockIdx.x * blockDim.x + threadIdx.x;
  if (i < n) p[i] = 0.0f;
}

// W [K=128 x N=128] fp32 -> transposed bf16, padded to WSTR: o[n*WSTR + k]
__global__ void k_prep_weight(const float* __restrict__ w, unsigned short* __restrict__ o) {
  int idx = blockIdx.x * blockDim.x + threadIdx.x;
  if (idx >= HH * WSTR) return;
  int n = idx / WSTR, k = idx % WSTR;
  o[idx] = (k < HH) ? f2bf(w[k * HH + n]) : (unsigned short)0;
}

// z[i] = (1 + eps[li]) * h[i]
__global__ void k_scale_init(const float* __restrict__ h, const float* __restrict__ eps, int li,
                             float* __restrict__ z, int n) {
  int i = blockIdx.x * blockDim.x + threadIdx.x;
  if (i < n) z[i] = (1.0f + eps[li]) * h[i];
}

// layer-0 (d=1) edge scatter
__global__ void k_edge_agg1(const float* __restrict__ h, const float* __restrict__ ew,
                            const int* __restrict__ src, const int* __restrict__ dst,
                            float* __restrict__ z, int e) {
  int i = blockIdx.x * blockDim.x + threadIdx.x;
  if (i < e) atomicAdd(&z[dst[i]], h[src[i]] * ew[i]);
}

// d=128 edge scatter: one thread per (edge, 4 channels)
__global__ void k_edge_agg128(const float* __restrict__ h, const float* __restrict__ ew,
                              const int* __restrict__ src, const int* __restrict__ dst,
                              float* __restrict__ z, int e) {
  int idx = blockIdx.x * blockDim.x + threadIdx.x;
  int total = e * 32;
  if (idx >= total) return;
  int ei = idx >> 5;
  int c  = (idx & 31) << 2;
  int s = src[ei], d = dst[ei];
  float w = ew[ei];
  const float4 hv = *reinterpret_cast<const float4*>(h + (size_t)s * HH + c);
  float* zp = z + (size_t)d * HH + c;
  atomicAdd(zp + 0, hv.x * w);
  atomicAdd(zp + 1, hv.y * w);
  atomicAdd(zp + 2, hv.z * w);
  atomicAdd(zp + 3, hv.w * w);
}

// layer-0 first "GEMM": outer product z0[n] * W1[1x128] + b1, with column stats
__global__ void k_outer_l0(const float* __restrict__ z0, const float* __restrict__ w1,
                           const float* __restrict__ b1, float* __restrict__ Y,
                           float* __restrict__ colStats, int rows) {
  __shared__ float csum[HH], csq[HH];
  if (threadIdx.x < HH) { csum[threadIdx.x] = 0.f; csq[threadIdx.x] = 0.f; }
  __syncthreads();
  long long total = (long long)rows * HH;
  for (long long idx = (long long)blockIdx.x * blockDim.x + threadIdx.x; idx < total;
       idx += (long long)gridDim.x * blockDim.x) {
    int r = (int)(idx >> 7);
    int c = (int)(idx & 127);
    float v = z0[r] * w1[c] + b1[c];
    Y[idx] = v;
    atomicAdd(&csum[c], v);
    atomicAdd(&csq[c], v * v);
  }
  __syncthreads();
  if (threadIdx.x < HH) {
    atomicAdd(&colStats[threadIdx.x],      csum[threadIdx.x]);
    atomicAdd(&colStats[HH + threadIdx.x], csq[threadIdx.x]);
  }
}

// fold column stats into BN affine: ss[c]=g*rsqrt(var+eps), ss[128+c]=b-ss[c]*mean
__global__ void k_bn_params(const float* __restrict__ cs, const float* __restrict__ g,
                            const float* __restrict__ b, float invN, float* __restrict__ ss) {
  int c = threadIdx.x;
  float m   = cs[c] * invN;
  float var = cs[HH + c] * invN - m * m;
  float a   = g[c] * rsqrtf(var + 1e-5f);
  ss[c]      = a;
  ss[HH + c] = b[c] - a * m;
}

// ---------------- WMMA GEMM: Y[rows x 128] = preop(X[rows x 128]) @ W[128 x 128] + bias -------------
// W comes pre-transposed + pre-padded (bf16, [N=128][WSTR]) and is staged to LDS with
// global_load_async_to_lds_b128 overlapped with the X-tile fp32->(BN affine,ReLU)->bf16 staging.
__global__ __launch_bounds__(256) void k_gemm128(
    const float* __restrict__ X, int rows,
    const float* __restrict__ ss, int relu,
    const unsigned short* __restrict__ Wt,    // bf16, [128][WSTR] (already W^T + padded)
    const float* __restrict__ bias,           // [128]
    float* __restrict__ Y,
    float* __restrict__ colStats)             // nullable [256]
{
  __shared__ __align__(16) unsigned short Xs[HH * WSTR];  // X tile, bf16, stride WSTR
  __shared__ __align__(16) unsigned short Ws[HH * WSTR];  // W^T image, async-DMA'd as-is
  __shared__ float csum[HH], csq[HH];

  const int t = threadIdx.x;
  if (t < HH) { csum[t] = 0.f; csq[t] = 0.f; }
  const int rowBase = blockIdx.x * 128;

  // ---- kick off async W DMA first (GVS: SGPR base + per-lane byte offset) ----
  {
    const unsigned int ldsW = (unsigned int)(uintptr_t)(&Ws[0]);
    const unsigned long long wAddr = (unsigned long long)(uintptr_t)Wt;
    const int nChunks = (HH * WSTR * 2) / 16;             // 2176 x 16B
    #pragma unroll
    for (int i = 0; i < 9; ++i) {
      int idx = t + i * 256;
      if (idx < nChunks) {
        unsigned int lo = ldsW + (unsigned int)(idx * 16);
        unsigned int go = (unsigned int)(idx * 16);
        asm volatile("global_load_async_to_lds_b128 %0, %1, %2"
                     :: "v"(lo), "v"(go), "s"(wAddr) : "memory");
      }
    }
  }

  __builtin_prefetch(bias, 0, 0);
  if (ss) __builtin_prefetch(ss, 0, 0);

  // ---- stage X tile (overlaps with async W DMA): fp32 -> affine -> relu -> bf16 ----
  #pragma unroll
  for (int i = 0; i < 16; ++i) {
    int idx = t + i * 256;                    // 0..4095 : 128 rows x 32 float4
    int r  = idx >> 5;
    int c4 = (idx & 31) << 2;
    int grow = rowBase + r;
    float4 v = make_float4(0.f, 0.f, 0.f, 0.f);
    if (grow < rows) v = *reinterpret_cast<const float4*>(X + (size_t)grow * HH + c4);
    if (ss) {
      v.x = v.x * ss[c4 + 0] + ss[HH + c4 + 0];
      v.y = v.y * ss[c4 + 1] + ss[HH + c4 + 1];
      v.z = v.z * ss[c4 + 2] + ss[HH + c4 + 2];
      v.w = v.w * ss[c4 + 3] + ss[HH + c4 + 3];
    }
    if (relu) {
      v.x = fmaxf(v.x, 0.f); v.y = fmaxf(v.y, 0.f);
      v.z = fmaxf(v.z, 0.f); v.w = fmaxf(v.w, 0.f);
    }
    unsigned int* d = reinterpret_cast<unsigned int*>(&Xs[r * WSTR + c4]);
    d[0] = pack2bf(v.x, v.y);
    d[1] = pack2bf(v.z, v.w);
  }

  asm volatile("s_wait_asynccnt 0x0" ::: "memory");   // my wave's DMA chunks are in LDS
  __syncthreads();                                     // everyone's chunks + X tile visible

  const int lane = t & 31;
  const int wid  = t >> 5;                    // 8 waves; wave owns rows [16*wid, 16*wid+16)
  const int half = lane >> 4;                 // K sub-group select per ISA 16-bit layout
  const int mrow = (wid << 4) + (lane & 15);  // A row this lane serves
  const int kgrp = half << 3;                 // +0 / +8
  const int ncol = lane & 15;

  v8f acc[8];
  #pragma unroll
  for (int nt = 0; nt < 8; ++nt) acc[nt] = (v8f)(0.0f);

  const unsigned short* aRow = &Xs[mrow * WSTR];
  #pragma unroll
  for (int ks = 0; ks < 4; ++ks) {            // K = 128 in steps of 32
    const int k0 = ks << 5;
    // A fragment: two b128 LDS loads (K contiguous: VGPR0-3 = k..k+7, VGPR4-7 = k+16..k+23)
    uint4 alo = *reinterpret_cast<const uint4*>(aRow + k0 + kgrp);
    uint4 ahi = *reinterpret_cast<const uint4*>(aRow + k0 + kgrp + 16);
    v8u abits = {alo.x, alo.y, alo.z, alo.w, ahi.x, ahi.y, ahi.z, ahi.w};
    v16bf afrag = __builtin_bit_cast(v16bf, abits);
    #pragma unroll
    for (int nt = 0; nt < 8; ++nt) {
      const unsigned short* bRow = &Ws[((nt << 4) + ncol) * WSTR];
      uint4 blo = *reinterpret_cast<const uint4*>(bRow + k0 + kgrp);
      uint4 bhi = *reinterpret_cast<const uint4*>(bRow + k0 + kgrp + 16);
      v8u bbits = {blo.x, blo.y, blo.z, blo.w, bhi.x, bhi.y, bhi.z, bhi.w};
      v16bf bfrag = __builtin_bit_cast(v16bf, bbits);
      acc[nt] = __builtin_amdgcn_wmma_f32_16x16x32_bf16(
          false, afrag, false, bfrag, (short)0, acc[nt], false, false);
    }
  }

  // epilogue: D layout -> lane holds rows (half*8 + r), col (nt*16 + ncol)
  const int rbase = rowBase + (wid << 4) + (half << 3);
  #pragma unroll
  for (int nt = 0; nt < 8; ++nt) {
    const int col = (nt << 4) + ncol;
    const float bb = bias[col];
    float s = 0.f, sq = 0.f;
    #pragma unroll
    for (int r = 0; r < 8; ++r) {
      int grow = rbase + r;
      if (grow < rows) {
        float val = acc[nt][r] + bb;
        Y[(size_t)grow * HH + col] = val;
        s  += val;
        sq += val * val;
      }
    }
    if (colStats) { atomicAdd(&csum[col], s); atomicAdd(&csq[col], sq); }
  }
  if (colStats) {
    __syncthreads();
    if (t < HH) {
      atomicAdd(&colStats[t],      csum[t]);
      atomicAdd(&colStats[HH + t], csq[t]);
    }
  }
}

// BN apply + ReLU -> Hout, fused sum-pooling into pooled[G x 128]
__global__ void k_bn_apply_pool(const float* __restrict__ Z, const float* __restrict__ ss,
                                const int* __restrict__ gid,
                                float* __restrict__ Hout, float* __restrict__ pooled, int rows) {
  int idx = blockIdx.x * blockDim.x + threadIdx.x;
  int total = rows * 32;
  if (idx >= total) return;
  int r  = idx >> 5;
  int c4 = (idx & 31) << 2;
  float4 z = *reinterpret_cast<const float4*>(Z + (size_t)r * HH + c4);
  float4 h;
  h.x = fmaxf(z.x * ss[c4 + 0] + ss[HH + c4 + 0], 0.f);
  h.y = fmaxf(z.y * ss[c4 + 1] + ss[HH + c4 + 1], 0.f);
  h.z = fmaxf(z.z * ss[c4 + 2] + ss[HH + c4 + 2], 0.f);
  h.w = fmaxf(z.w * ss[c4 + 3] + ss[HH + c4 + 3], 0.f);
  *reinterpret_cast<float4*>(Hout + (size_t)r * HH + c4) = h;
  int g = gid[r];
  float* pp = pooled + (size_t)g * HH + c4;
  atomicAdd(pp + 0, h.x);
  atomicAdd(pp + 1, h.y);
  atomicAdd(pp + 2, h.z);
  atomicAdd(pp + 3, h.w);
}

// score += pooled[Gx128] @ predW[128x7] + predB
__global__ void k_score_add(const float* __restrict__ pooled, const float* __restrict__ pw,
                            const float* __restrict__ pb, float* __restrict__ score) {
  int idx = blockIdx.x * blockDim.x + threadIdx.x;
  if (idx >= GG * OUTD) return;
  int g = idx / OUTD, o = idx % OUTD;
  float s = pb[o];
  const float* pr = pooled + (size_t)g * HH;
  #pragma unroll 8
  for (int c = 0; c < HH; ++c) s += pr[c] * pw[c * OUTD + o];
  score[idx] += s;
}

// rep0 pooling (features are [N,1])
__global__ void k_pool1(const float* __restrict__ f, const int* __restrict__ gid,
                        float* __restrict__ pooled0, int n) {
  int i = blockIdx.x * blockDim.x + threadIdx.x;
  if (i < n) atomicAdd(&pooled0[gid[i]], f[i]);
}
__global__ void k_score1(const float* __restrict__ pooled0, const float* __restrict__ pw,
                         const float* __restrict__ pb, float* __restrict__ score) {
  int idx = blockIdx.x * blockDim.x + threadIdx.x;
  if (idx >= GG * OUTD) return;
  int g = idx / OUTD, o = idx % OUTD;
  score[idx] += pooled0[g] * pw[o] + pb[o];
}

// ---------------- orchestration ----------------
extern "C" void kernel_launch(void* const* d_in, const int* in_sizes, int n_in,
                              void* d_out, int out_size, void* d_ws, size_t ws_size,
                              hipStream_t stream) {
  (void)in_sizes; (void)n_in; (void)out_size; (void)ws_size;
  const float* features = (const float*)d_in[0];
  const float* ew       = (const float*)d_in[1];
  const float* eps      = (const float*)d_in[2];
  // layers: 8 arrays each {W1,b1,bn1_g,bn1_b,W2,b2,bn_g,bn_b}
  auto Lp = [&](int i, int f) -> const float* { return (const float*)d_in[3 + i * 8 + f]; };
  // pred: (L+1) x {W, b}
  auto Pp = [&](int i, int f) -> const float* { return (const float*)d_in[3 + 8 * LL + i * 2 + f]; };
  const int* src = (const int*)d_in[3 + 8 * LL + 2 * (LL + 1) + 0];
  const int* dst = (const int*)d_in[3 + 8 * LL + 2 * (LL + 1) + 1];
  const int* gid = (const int*)d_in[3 + 8 * LL + 2 * (LL + 1) + 2];
  float* out = (float*)d_out;

  // workspace layout
  const size_t WSLOT = (size_t)HH * WSTR;          // halfwords per prepped weight matrix
  char* wsb = (char*)d_ws;
  float* bufA = (float*)wsb;                       // [N x 128]
  float* bufB = bufA + (size_t)NN * HH;            // [N x 128]
  float* z0   = bufB + (size_t)NN * HH;            // [N]
  unsigned short* Wb = (unsigned short*)(z0 + NN); // 23 prepped bf16 matrices
  float* colStats = (float*)(Wb + 23 * WSLOT);     // [256]
  float* ssbuf    = colStats + 256;                // [256]
  float* pooled   = ssbuf + 256;                   // [G x 128]
  float* pooled0  = pooled + (size_t)GG * HH;      // [G]

  const int T = 256;
  auto cdiv = [](long long a, long long b) { return (unsigned int)((a + b - 1) / b); };
  const float invN = 1.0f / (float)NN;

  // prep weights: transpose + bf16 + pad-to-WSTR. W2_i at slot i; W1_i (i>=1) at slot 11+i
  for (int i = 0; i < LL; ++i)
    k_prep_weight<<<cdiv(HH * WSTR, T), T, 0, stream>>>(Lp(i, 4), Wb + (size_t)i * WSLOT);
  for (int i = 1; i < LL; ++i)
    k_prep_weight<<<cdiv(HH * WSTR, T), T, 0, stream>>>(Lp(i, 0), Wb + (size_t)(11 + i) * WSLOT);

  // score init + rep0 readout
  k_fill_zero<<<cdiv(GG * OUTD, T), T, 0, stream>>>(out, GG * OUTD);
  k_fill_zero<<<cdiv(GG, T), T, 0, stream>>>(pooled0, GG);
  k_pool1<<<cdiv(NN, T), T, 0, stream>>>(features, gid, pooled0, NN);
  k_score1<<<cdiv(GG * OUTD, T), T, 0, stream>>>(pooled0, Pp(0, 0), Pp(0, 1), out);

  // ---- layer 0 (d_in = 1) ----
  k_scale_init<<<cdiv(NN, T), T, 0, stream>>>(features, eps, 0, z0, NN);
  k_edge_agg1<<<cdiv(EE, T), T, 0, stream>>>(features, ew, src, dst, z0, EE);
  k_fill_zero<<<1, 256, 0, stream>>>(colStats, 256);
  k_outer_l0<<<2048, T, 0, stream>>>(z0, Lp(0, 0), Lp(0, 1), bufA, colStats, NN);
  k_bn_params<<<1, HH, 0, stream>>>(colStats, Lp(0, 2), Lp(0, 3), invN, ssbuf);
  k_fill_zero<<<1, 256, 0, stream>>>(colStats, 256);
  k_gemm128<<<cdiv(NN, 128), T, 0, stream>>>(bufA, NN, ssbuf, 1, Wb + (size_t)0 * WSLOT,
                                             Lp(0, 5), bufB, colStats);
  k_bn_params<<<1, HH, 0, stream>>>(colStats, Lp(0, 6), Lp(0, 7), invN, ssbuf);
  k_fill_zero<<<cdiv(GG * HH, T), T, 0, stream>>>(pooled, GG * HH);
  k_bn_apply_pool<<<cdiv((long long)NN * 32, T), T, 0, stream>>>(bufB, ssbuf, gid, bufA, pooled, NN);
  k_score_add<<<cdiv(GG * OUTD, T), T, 0, stream>>>(pooled, Pp(1, 0), Pp(1, 1), out);

  // ---- layers 1..11 (h lives in bufA) ----
  for (int i = 1; i < LL; ++i) {
    k_scale_init<<<cdiv((long long)NN * HH, T), T, 0, stream>>>(bufA, eps, i, bufB, NN * HH);
    k_edge_agg128<<<cdiv((long long)EE * 32, T), T, 0, stream>>>(bufA, ew, src, dst, bufB, EE);
    k_fill_zero<<<1, 256, 0, stream>>>(colStats, 256);
    k_gemm128<<<cdiv(NN, 128), T, 0, stream>>>(bufB, NN, nullptr, 0, Wb + (size_t)(11 + i) * WSLOT,
                                               Lp(i, 1), bufA, colStats);
    k_bn_params<<<1, HH, 0, stream>>>(colStats, Lp(i, 2), Lp(i, 3), invN, ssbuf);
    k_fill_zero<<<1, 256, 0, stream>>>(colStats, 256);
    k_gemm128<<<cdiv(NN, 128), T, 0, stream>>>(bufA, NN, ssbuf, 1, Wb + (size_t)i * WSLOT,
                                               Lp(i, 5), bufB, colStats);
    k_bn_params<<<1, HH, 0, stream>>>(colStats, Lp(i, 6), Lp(i, 7), invN, ssbuf);
    k_fill_zero<<<cdiv(GG * HH, T), T, 0, stream>>>(pooled, GG * HH);
    k_bn_apply_pool<<<cdiv((long long)NN * 32, T), T, 0, stream>>>(bufB, ssbuf, gid, bufA, pooled, NN);
    k_score_add<<<cdiv(GG * OUTD, T), T, 0, stream>>>(pooled, Pp(i + 1, 0), Pp(i + 1, 1), out);
  }
}
